// VanillaLSTM_32822140076725
// MI455X (gfx1250) — matrix operand order repaired and checked by reading, not compiled, not run on hardware
//
#include <hip/hip_runtime.h>
#include <hip/hip_bf16.h>

typedef __attribute__((ext_vector_type(16))) _Float16 v16h;
typedef __attribute__((ext_vector_type(8)))  float    v8f;
typedef int v4i __attribute__((vector_size(16)));

#define BATCH 8192
#define TSTEPS 128
#define HID 128
#define KCAT 256
#define EPSBN 1e-5f

#if defined(__gfx1250__) && __has_builtin(__builtin_amdgcn_global_load_async_to_lds_b128) && __has_builtin(__builtin_amdgcn_s_wait_asynccnt)
#define USE_ASYNC_LDS 1
#endif

// ---------------------------------------------------------------------------
// Setup: zero state, build fp16 concat weight [512][256], fused bias, zero BN stats
// ---------------------------------------------------------------------------
__global__ __launch_bounds__(256) void lstm_setup_kernel(
    const float* __restrict__ Wih, const float* __restrict__ Whh,
    const float* __restrict__ bih, const float* __restrict__ bhh,
    _Float16* __restrict__ Wcat16, float* __restrict__ biascat,
    float* __restrict__ hA, float* __restrict__ hB, float* __restrict__ cbuf,
    float* __restrict__ stats)
{
    int idx = blockIdx.x * 256 + threadIdx.x;           // 1,048,576 threads
    hA[idx] = 0.f; hB[idx] = 0.f; cbuf[idx] = 0.f;
    if (idx < 512 * 256) {
        int g = idx >> 8, k = idx & 255;
        float v = (k < HID) ? Wih[g * HID + k] : Whh[g * HID + (k - HID)];
        Wcat16[idx] = (_Float16)v;
    }
    if (idx < 512)  biascat[idx] = bih[idx] + bhh[idx];
    if (idx < 1280) stats[idx] = 0.f;
}

// ---------------------------------------------------------------------------
// Embed prep: per-timestep batch moments of x -> per-feature affine coefs
// e[b,j] = relu(cA[j]*x0 + cB[j]*x1 + cC[j])   (BN folded; b_emb cancels)
// ---------------------------------------------------------------------------
__global__ __launch_bounds__(256) void embed_prep_kernel(
    const float* __restrict__ xin, const float* __restrict__ Wemb,
    const float* __restrict__ gamma, const float* __restrict__ beta,
    float* __restrict__ cA, float* __restrict__ cB, float* __restrict__ cC)
{
    __shared__ float red[5][256];
    int t = blockIdx.x, tid = threadIdx.x;
    float s0 = 0.f, s1 = 0.f, s00 = 0.f, s11 = 0.f, s01 = 0.f;
    for (int b = tid; b < BATCH; b += 256) {
        float x0 = xin[(size_t)b * (TSTEPS * 2) + t * 2 + 0];
        float x1 = xin[(size_t)b * (TSTEPS * 2) + t * 2 + 1];
        s0 += x0; s1 += x1; s00 += x0 * x0; s11 += x1 * x1; s01 += x0 * x1;
    }
    red[0][tid] = s0; red[1][tid] = s1; red[2][tid] = s00;
    red[3][tid] = s11; red[4][tid] = s01;
    __syncthreads();
    for (int s = 128; s > 0; s >>= 1) {
        if (tid < s)
            for (int k = 0; k < 5; ++k) red[k][tid] += red[k][tid + s];
        __syncthreads();
    }
    if (tid < HID) {
        const float invB = 1.f / (float)BATCH;
        float m0 = red[0][0] * invB, m1 = red[1][0] * invB;
        float c00 = red[2][0] * invB - m0 * m0;
        float c11 = red[3][0] * invB - m1 * m1;
        float c01 = red[4][0] * invB - m0 * m1;
        float w0 = Wemb[tid * 2 + 0], w1 = Wemb[tid * 2 + 1];
        float varY = c00 * w0 * w0 + 2.f * c01 * w0 * w1 + c11 * w1 * w1;
        float ghat = gamma[tid] * rsqrtf(varY + EPSBN);
        float a = ghat * w0, b = ghat * w1;
        cA[t * HID + tid] = a;
        cB[t * HID + tid] = b;
        cC[t * HID + tid] = beta[tid] - a * m0 - b * m1;
    }
}

// ---------------------------------------------------------------------------
// Per-timestep fused: embed epilogue -> WMMA GEMM (gates) -> LSTM cell update
// grid (128, 4): 64 rows x (4 gates x 32 units) per block, 8 waves,
// double-buffered LDS chunks, async global->LDS weight staging when available
// ---------------------------------------------------------------------------
__global__ __launch_bounds__(256) void lstm_step_kernel(
    const float* __restrict__ xin,
    const _Float16* __restrict__ Wcat16,   // [512][256]
    const float* __restrict__ biascat,     // [512]
    const float* __restrict__ cA, const float* __restrict__ cB,
    const float* __restrict__ cC,
    const float* __restrict__ hin, float* __restrict__ hout,
    float* __restrict__ cbuf, int t)
{
    __shared__ __align__(16) _Float16 As16[2][64 * 32];     // A chunks (dbl buf)
    __shared__ __align__(16) uint4    Bs4[2][512];          // B chunks (dbl buf)
    __shared__ float Gt[64 * 128];                          // gate exchange
    __shared__ float xs[64 * 2];

    const int tid  = threadIdx.x;
    const int lane = tid & 31;
    const int wave = tid >> 5;
    const int mhalf = wave >> 2;          // 0/1 -> rows 0-31 / 32-63
    const int gate  = wave & 3;           // i,f,g,o
    const int b0 = blockIdx.x * 64;
    const int nblk = blockIdx.y;          // 32 hidden units per block

    const float* tcA = cA + t * HID;
    const float* tcB = cB + t * HID;
    const float* tcC = cC + t * HID;
    const uint4* Wv = reinterpret_cast<const uint4*>(Wcat16);

    if (tid < 128) {
        int r = tid >> 1, i = tid & 1;
        xs[tid] = xin[(size_t)(b0 + r) * (TSTEPS * 2) + t * 2 + i];
    }
    __syncthreads();                      // xs visible before e staging

    // stage one 32-wide K chunk into LDS buffer `buf`
    auto stageB = [&](int kc, int buf) {
        for (int i = tid; i < 512; i += 256) {
            int row = i >> 2, part = i & 3;
            int w = (row >> 5) * HID + nblk * 32 + (row & 31);  // gate*128+col
            const uint4* src = Wv + (size_t)w * 32 + kc * 4 + part;
#if USE_ASYNC_LDS
            __builtin_amdgcn_global_load_async_to_lds_b128(
                (v4i*)src, (v4i*)&Bs4[buf][i], 0, 0);
#else
            Bs4[buf][i] = *src;
#endif
        }
    };
    auto stageA = [&](int kc, int buf) {
        const int kbase = kc * 32;
        if (kc < 4) {
            for (int i = tid; i < 2048; i += 256) {
                int r = i >> 5, jj = i & 31, j = kbase + jj;
                float e = tcA[j] * xs[r * 2] + tcB[j] * xs[r * 2 + 1] + tcC[j];
                As16[buf][r * 32 + jj] = (_Float16)(e > 0.f ? e : 0.f);
            }
        } else {
            for (int i = tid; i < 2048; i += 256) {
                int r = i >> 5, jj = i & 31, k = kbase - HID + jj;
                As16[buf][r * 32 + jj] = (_Float16)hin[(size_t)(b0 + r) * HID + k];
            }
        }
    };

    v8f acc[2][2] = {};

    stageB(0, 0);
    stageA(0, 0);
#if USE_ASYNC_LDS
    __builtin_amdgcn_s_wait_asynccnt(0);
#endif
    __syncthreads();

    for (int kc = 0; kc < 8; ++kc) {
        const int cur = kc & 1;
        if (kc < 7) {                       // prefetch next chunk under WMMA
            stageB(kc + 1, cur ^ 1);
            stageA(kc + 1, cur ^ 1);
        }

        const unsigned int* AsU = reinterpret_cast<const unsigned int*>(As16[cur]);
        const unsigned int* BsU = reinterpret_cast<const unsigned int*>(Bs4[cur]);

        // fragments per ISA 7.12.2, then 4x v_wmma_f32_16x16x32_f16
        v16h afr[2], bfr[2];
        #pragma unroll
        for (int mi = 0; mi < 2; ++mi) {
            int r = mhalf * 32 + mi * 16 + (lane & 15);
            unsigned int* au = reinterpret_cast<unsigned int*>(&afr[mi]);
            #pragma unroll
            for (int v = 0; v < 8; ++v)
                au[v] = AsU[r * 16 + (v >> 2) * 8 + (lane >> 4) * 4 + (v & 3)];
        }
        #pragma unroll
        for (int ni = 0; ni < 2; ++ni) {
            int n = gate * 32 + ni * 16 + (lane & 15);
            unsigned int* bu = reinterpret_cast<unsigned int*>(&bfr[ni]);
            #pragma unroll
            for (int v = 0; v < 8; ++v)
                bu[v] = BsU[n * 16 + (v >> 2) * 8 + (lane >> 4) * 4 + (v & 3)];
        }
        #pragma unroll
        for (int mi = 0; mi < 2; ++mi)
            #pragma unroll
            for (int ni = 0; ni < 2; ++ni)
                acc[mi][ni] = __builtin_amdgcn_wmma_f32_16x16x32_f16(
                    false, afr[mi], false, bfr[ni], (short)0, acc[mi][ni],
                    false, false);

#if USE_ASYNC_LDS
        __builtin_amdgcn_s_wait_asynccnt(0);
#endif
        __syncthreads();                    // next chunk staged & frags consumed
    }

    // exchange gate tiles through LDS (D layout: vgpr r -> M, lane -> N)
    #pragma unroll
    for (int mi = 0; mi < 2; ++mi)
        #pragma unroll
        for (int ni = 0; ni < 2; ++ni) {
            int mb = mhalf * 32 + mi * 16 + (lane >> 4) * 8;
            int nb = gate * 32 + ni * 16 + (lane & 15);
            #pragma unroll
            for (int r = 0; r < 8; ++r)
                Gt[(mb + r) * 128 + nb] = acc[mi][ni][r];
        }
    __syncthreads();

    // fused LSTM cell update: 64 rows x 32 units
    for (int i = tid; i < 2048; i += 256) {
        int r = i >> 5, u = i & 31;
        int gi = nblk * 32 + u;                       // hidden unit index
        float iv = Gt[r * 128 + 0 * 32 + u] + biascat[0 * HID + gi];
        float fv = Gt[r * 128 + 1 * 32 + u] + biascat[1 * HID + gi];
        float gv = Gt[r * 128 + 2 * 32 + u] + biascat[2 * HID + gi];
        float ov = Gt[r * 128 + 3 * 32 + u] + biascat[3 * HID + gi];
        float ig = 1.f / (1.f + __expf(-iv));
        float fg = 1.f / (1.f + __expf(-fv));
        float gg = tanhf(gv);
        float og = 1.f / (1.f + __expf(-ov));
        size_t idx = (size_t)(b0 + r) * HID + gi;
        float cn = fg * cbuf[idx] + ig * gg;
        cbuf[idx] = cn;
        hout[idx] = og * tanhf(cn);
    }
}

// ---------------------------------------------------------------------------
// Output projection (128 -> 5) + BN moment accumulation
// ---------------------------------------------------------------------------
__global__ __launch_bounds__(256) void outproj_kernel(
    const float* __restrict__ h, const float* __restrict__ Wout,
    const float* __restrict__ bout, float* __restrict__ z,
    float* __restrict__ stats, int t)
{
    __shared__ float Ws[5 * 128];
    __shared__ float bs[5];
    __shared__ float ls[10];
    int tid = threadIdx.x;
    for (int i = tid; i < 640; i += 256) Ws[i] = Wout[i];
    if (tid < 5)  bs[tid] = bout[tid];
    if (tid < 10) ls[tid] = 0.f;
    __syncthreads();

    int b = blockIdx.x * 256 + tid;
    float zz[5];
    #pragma unroll
    for (int d = 0; d < 5; ++d) {
        float a = bs[d];
        for (int k = 0; k < HID; ++k)
            a += h[(size_t)b * HID + k] * Ws[d * HID + k];
        zz[d] = a;
        z[b * 5 + d] = a;
    }
    #pragma unroll
    for (int d = 0; d < 5; ++d) {
        atomicAdd(&ls[d], zz[d]);
        atomicAdd(&ls[5 + d], zz[d] * zz[d]);
    }
    __syncthreads();
    if (tid < 10) atomicAdd(&stats[t * 10 + tid], ls[tid]);
}

// ---------------------------------------------------------------------------
// Output BN normalize -> outs[B, T, 5]
// ---------------------------------------------------------------------------
__global__ __launch_bounds__(256) void outnorm_kernel(
    const float* __restrict__ z, const float* __restrict__ stats,
    const float* __restrict__ gamma, const float* __restrict__ beta,
    float* __restrict__ out, int t)
{
    int b = blockIdx.x * 256 + threadIdx.x;
    const float invB = 1.f / (float)BATCH;
    #pragma unroll
    for (int d = 0; d < 5; ++d) {
        float mu  = stats[t * 10 + d] * invB;
        float var = stats[t * 10 + 5 + d] * invB - mu * mu;
        float v = z[b * 5 + d];
        out[(size_t)b * (TSTEPS * 5) + t * 5 + d] =
            gamma[d] * (v - mu) * rsqrtf(var + EPSBN) + beta[d];
    }
}

// ---------------------------------------------------------------------------
// Emit final (h, c)
// ---------------------------------------------------------------------------
__global__ __launch_bounds__(256) void finalize_kernel(
    const float* __restrict__ hfin, const float* __restrict__ cbuf,
    float* __restrict__ out)
{
    size_t idx = (size_t)blockIdx.x * 256 + threadIdx.x;   // 1,048,576
    const size_t OFF = (size_t)BATCH * TSTEPS * 5;
    out[OFF + idx] = hfin[idx];
    out[OFF + (size_t)BATCH * HID + idx] = cbuf[idx];
}

extern "C" void kernel_launch(void* const* d_in, const int* in_sizes, int n_in,
                              void* d_out, int out_size, void* d_ws, size_t ws_size,
                              hipStream_t stream) {
    const float* xin   = (const float*)d_in[0];
    const float* Wemb  = (const float*)d_in[1];
    // d_in[2] = b_emb (cancels inside BN)
    const float* gemb  = (const float*)d_in[3];
    const float* bbeta = (const float*)d_in[4];
    const float* Wih   = (const float*)d_in[5];
    const float* bih   = (const float*)d_in[6];
    const float* Whh   = (const float*)d_in[7];
    const float* bhh   = (const float*)d_in[8];
    const float* Wout  = (const float*)d_in[9];
    const float* bout  = (const float*)d_in[10];
    const float* gout  = (const float*)d_in[11];
    const float* bout2 = (const float*)d_in[12];
    float* out = (float*)d_out;

    char* ws = (char*)d_ws;
    _Float16* Wcat16 = (_Float16*)(ws + 0);                  // 256 KB
    float* biascat   = (float*)(ws + 262144);                // 2 KB
    float* cA        = (float*)(ws + 264192);                // 64 KB
    float* cB        = (float*)(ws + 329728);                // 64 KB
    float* cC        = (float*)(ws + 395264);                // 64 KB
    float* hA        = (float*)(ws + 460800);                // 4 MB
    float* hB        = (float*)(ws + 460800 + 4194304);      // 4 MB
    float* cbuf      = (float*)(ws + 460800 + 2 * 4194304);  // 4 MB
    float* zbuf      = (float*)(ws + 460800 + 3 * 4194304);  // 160 KB
    float* stats     = (float*)(ws + 460800 + 3 * 4194304 + 163840);

    lstm_setup_kernel<<<4096, 256, 0, stream>>>(Wih, Whh, bih, bhh, Wcat16,
                                                biascat, hA, hB, cbuf, stats);
    embed_prep_kernel<<<128, 256, 0, stream>>>(xin, Wemb, gemb, bbeta, cA, cB, cC);

    for (int t = 0; t < TSTEPS; ++t) {
        const float* hin = (t & 1) ? hB : hA;
        float* hout      = (t & 1) ? hA : hB;
        lstm_step_kernel<<<dim3(128, 4), 256, 0, stream>>>(
            xin, Wcat16, biascat, cA, cB, cC, hin, hout, cbuf, t);
        outproj_kernel<<<32, 256, 0, stream>>>(hout, Wout, bout, zbuf, stats, t);
        outnorm_kernel<<<32, 256, 0, stream>>>(zbuf, stats, gout, bout2, out, t);
    }
    // t = 127 is odd -> final h lives in hA
    finalize_kernel<<<4096, 256, 0, stream>>>(hA, cbuf, out);
}
